// MiMoMoeDecoderLayer_68753836474422
// MI455X (gfx1250) — matrix operand order repaired and checked by reading, not compiled, not run on hardware
//
#include <hip/hip_runtime.h>
#include <hip/hip_bf16.h>

// MiMo MoE decoder layer for gfx1250 (MI455X).
// Heavy math on v_wmma_f32_16x16x32_bf16; A-tile / K-tile staging via the
// Tensor Data Mover (tensor_load_to_lds + s_wait_tensorcnt) when available.

typedef __attribute__((ext_vector_type(16))) __bf16       v16bf;
typedef __attribute__((ext_vector_type(8)))  __bf16       v8bf;
typedef __attribute__((ext_vector_type(8)))  float        v8f;
typedef __attribute__((ext_vector_type(4)))  unsigned int v4ui;
typedef __attribute__((ext_vector_type(8)))  int          v8i;
typedef __attribute__((ext_vector_type(4)))  int          v4i;

constexpr int Td   = 2048;
constexpr int Dd   = 2048;
constexpr int Hh   = 16;
constexpr int KVHh = 4;
constexpr int HDd  = 128;
constexpr int Ee   = 8;
constexpr int Ii   = 1024;

#if defined(__HIP_DEVICE_COMPILE__) && __has_builtin(__builtin_amdgcn_tensor_load_to_lds)
#define USE_TDM 1
#else
#define USE_TDM 0
#endif

static __device__ __forceinline__ v8f wmma_bf16(v16bf a, v16bf b, v8f c) {
  return __builtin_amdgcn_wmma_f32_16x16x32_bf16(false, a, false, b, (short)0, c,
                                                 false, false);
}
static __device__ __forceinline__ v8bf ld8(const __bf16* p) {
  return *reinterpret_cast<const v8bf*>(p);
}

#if USE_TDM
// Issue a TDM 2-D tile load: tile_x contiguous bf16 elems per row, tile_y rows,
// row pitch stride_elems, into LDS at lds_off (contiguous row-major fill).
static __device__ __forceinline__ void tdm_load_2d(unsigned int lds_off,
                                                   const void* gptr,
                                                   unsigned int rowlen_elems,
                                                   unsigned long long stride_elems,
                                                   unsigned int tile_x,
                                                   unsigned int tile_y) {
  unsigned long long ga = (unsigned long long)gptr;
  v4ui g0;
  g0[0] = 1u;                                   // count=1 (valid user descriptor)
  g0[1] = lds_off;                              // D# lds_addr
  g0[2] = (unsigned int)(ga & 0xffffffffu);     // global_addr[31:0]
  g0[3] = (unsigned int)((ga >> 32) & 0x01ffffffu) | 0x80000000u;  // [56:32]|type=2
  unsigned int td0 = rowlen_elems;
  unsigned int td1 = 1u << 20;                  // large tensor_dim1 (no OOB clip)
  v8i g1;
  g1[0] = (int)(1u << 16);                      // data_size = 1 (2 bytes)
  g1[1] = (int)((td0 & 0xffffu) << 16);         // tensor_dim0[15:0]  -> bits 63:48
  g1[2] = (int)((td0 >> 16) | ((td1 & 0xffffu) << 16));  // dim0 hi | dim1 lo
  g1[3] = (int)((td1 >> 16) | (tile_x << 16));  // dim1 hi | tile_dim0
  g1[4] = (int)(tile_y & 0xffffu);              // tile_dim1 (tile_dim2 = 0)
  g1[5] = (int)(stride_elems & 0xffffffffu);    // tensor_dim0_stride[31:0]
  g1[6] = (int)((stride_elems >> 32) & 0xffffu);// tensor_dim0_stride[47:32]
  g1[7] = 0;
  v4i z4 = {0, 0, 0, 0};
#if defined(__clang_major__) && __clang_major__ >= 23
  v8i z8 = {0, 0, 0, 0, 0, 0, 0, 0};
  __builtin_amdgcn_tensor_load_to_lds(g0, g1, z4, z4, z8, 0);
#else
  __builtin_amdgcn_tensor_load_to_lds(g0, g1, z4, z4, 0);
#endif
}
static __device__ __forceinline__ unsigned int lds_offset(const void* p) {
  return (unsigned int)(unsigned long long)p;   // low 32 bits = LDS byte offset
}
#endif

// reductions across a 16-lane half of a wave32 (rows live per-half)
static __device__ __forceinline__ float rmax_half(float x) {
  x = fmaxf(x, __shfl_xor(x, 1, 32));
  x = fmaxf(x, __shfl_xor(x, 2, 32));
  x = fmaxf(x, __shfl_xor(x, 4, 32));
  x = fmaxf(x, __shfl_xor(x, 8, 32));
  return x;
}
static __device__ __forceinline__ float rsum_half(float x) {
  x += __shfl_xor(x, 1, 32);
  x += __shfl_xor(x, 2, 32);
  x += __shfl_xor(x, 4, 32);
  x += __shfl_xor(x, 8, 32);
  return x;
}

__global__ void cast_bf16_kernel(const float* __restrict__ in, __bf16* __restrict__ o,
                                 long n) {
  long i = (long)blockIdx.x * blockDim.x + threadIdx.x;
  long stride = (long)gridDim.x * blockDim.x;
  for (; i < n; i += stride) o[i] = (__bf16)in[i];
}

__global__ __launch_bounds__(256) void rmsnorm_bf16_kernel(const float* __restrict__ x,
                                                           const float* __restrict__ w,
                                                           __bf16* __restrict__ o) {
  __shared__ float red[256];
  int row = blockIdx.x;
  const float* xr = x + (size_t)row * Dd;
  float ss = 0.f;
  for (int i = threadIdx.x; i < Dd; i += 256) { float v = xr[i]; ss += v * v; }
  red[threadIdx.x] = ss;
  __syncthreads();
  for (int s = 128; s > 0; s >>= 1) {
    if (threadIdx.x < s) red[threadIdx.x] += red[threadIdx.x + s];
    __syncthreads();
  }
  float rr = rsqrtf(red[0] / (float)Dd + 1e-6f);
  for (int i = threadIdx.x; i < Dd; i += 256)
    o[(size_t)row * Dd + i] = (__bf16)(xr[i] * rr * w[i]);
}

__global__ void rope_bf16_kernel(const float* __restrict__ in, const int* __restrict__ pos,
                                 __bf16* __restrict__ o, int heads, long total) {
  long i = (long)blockIdx.x * blockDim.x + threadIdx.x;
  if (i >= total) return;
  int j  = (int)(i & 63);
  int hh = (int)((i >> 6) % heads);
  int t  = (int)(i / (64L * heads));
  float inv = __expf(-(float)j * (13.815510557964274f / 64.f));  // theta^(-j/64)
  float fr = (float)pos[t] * inv;
  float c = __cosf(fr), s = __sinf(fr);
  const float* base = in + ((size_t)t * heads + hh) * HDd;
  float x1 = base[j], x2 = base[j + 64];
  __bf16* ob = o + ((size_t)t * heads + hh) * HDd;
  ob[j]      = (__bf16)(x1 * c - x2 * s);
  ob[j + 64] = (__bf16)(x2 * c + x1 * s);
}

// C[M,N] = A[M,K] @ B[K,N] (+bias[N]) (+resid[M,N]); A,B bf16 row-major, C f32.
// 256 threads = 8 waves; block tile 128x64, K-step 32; wave tile 32x32 (4 WMMA).
// A tile staged by the Tensor Data Mover; B staged transposed by the threads.
__global__ __launch_bounds__(256) void gemm_bf16_kernel(const __bf16* __restrict__ A,
                                                        const __bf16* __restrict__ B,
                                                        float* __restrict__ C,
                                                        const float* __restrict__ bias,
                                                        const float* __restrict__ resid,
                                                        int N, int Kd) {
  __shared__ __bf16 As[128][32];
  __shared__ __bf16 Bt[64][32];  // B transposed: [n][k]
  const int m0 = blockIdx.y * 128, n0 = blockIdx.x * 64;
  const int tid = threadIdx.x;
  const int lane = tid & 31, wv = tid >> 5;
  const int hf = lane >> 4, idx = lane & 15;
  const int wm = wv & 3, wn = wv >> 2;
  const int brow = tid >> 3, bcol = (tid & 7) * 8;
  v8f acc00 = {}, acc01 = {}, acc10 = {}, acc11 = {};
  for (int k0 = 0; k0 < Kd; k0 += 32) {
    __syncthreads();
#if USE_TDM
    if (wv == 0)
      tdm_load_2d(lds_offset(&As[0][0]), A + (size_t)m0 * Kd + k0,
                  (unsigned int)Kd, (unsigned long long)Kd, 32u, 128u);
#else
    {
      int ar = tid >> 1, ac = (tid & 1) * 16;
      *reinterpret_cast<v8bf*>(&As[ar][ac]) =
          ld8(&A[(size_t)(m0 + ar) * Kd + k0 + ac]);
      *reinterpret_cast<v8bf*>(&As[ar][ac + 8]) =
          ld8(&A[(size_t)(m0 + ar) * Kd + k0 + ac + 8]);
    }
#endif
    v8bf bvv = ld8(&B[(size_t)(k0 + brow) * N + n0 + bcol]);
#pragma unroll
    for (int j = 0; j < 8; ++j) Bt[bcol + j][brow] = bvv[j];
    if (k0 + 32 < Kd)
      __builtin_prefetch(&B[(size_t)(k0 + 32 + brow) * N + n0 + bcol], 0, 0);
#if USE_TDM
    if (wv == 0) __builtin_amdgcn_s_wait_tensorcnt(0);
#endif
    __syncthreads();
    // A frags (16x32): lane row = idx, elems 0-7 = K hf*8.., 8-15 = K 16+hf*8..
    v16bf af0, af1;
    {
      const __bf16* pa = &As[wm * 32 + idx][hf * 8];
      v8bf lo = ld8(pa), hi = ld8(pa + 16);
#pragma unroll
      for (int i = 0; i < 8; ++i) { af0[i] = lo[i]; af0[i + 8] = hi[i]; }
      pa = &As[wm * 32 + 16 + idx][hf * 8];
      lo = ld8(pa); hi = ld8(pa + 16);
#pragma unroll
      for (int i = 0; i < 8; ++i) { af1[i] = lo[i]; af1[i + 8] = hi[i]; }
    }
    // B frags (32x16): lane = col, elems 0-7 = K hf*8.., 8-15 = K 16+hf*8..
    v16bf bf0, bf1;
    {
      const __bf16* pb = &Bt[wn * 32 + idx][hf * 8];
      v8bf lo = ld8(pb), hi = ld8(pb + 16);
#pragma unroll
      for (int i = 0; i < 8; ++i) { bf0[i] = lo[i]; bf0[i + 8] = hi[i]; }
      pb = &Bt[wn * 32 + 16 + idx][hf * 8];
      lo = ld8(pb); hi = ld8(pb + 16);
#pragma unroll
      for (int i = 0; i < 8; ++i) { bf1[i] = lo[i]; bf1[i + 8] = hi[i]; }
    }
    acc00 = wmma_bf16(af0, bf0, acc00);
    acc01 = wmma_bf16(af0, bf1, acc01);
    acc10 = wmma_bf16(af1, bf0, acc10);
    acc11 = wmma_bf16(af1, bf1, acc11);
  }
  // C layout: elem v, lane: row = v + hf*8, col = idx
#pragma unroll
  for (int rt = 0; rt < 2; ++rt) {
    v8f a0 = rt ? acc10 : acc00;
    v8f a1 = rt ? acc11 : acc01;
#pragma unroll
    for (int v = 0; v < 8; ++v) {
      int r = m0 + wm * 32 + rt * 16 + v + hf * 8;
      int c0 = n0 + wn * 32 + idx;
      int c1 = c0 + 16;
      float v0 = a0[v], v1 = a1[v];
      if (bias)  { v0 += bias[c0]; v1 += bias[c1]; }
      if (resid) { v0 += resid[(size_t)r * N + c0]; v1 += resid[(size_t)r * N + c1]; }
      C[(size_t)r * N + c0] = v0;
      C[(size_t)r * N + c1] = v1;
    }
  }
}

// Flash attention, causal, GQA (head h uses kv head h/4).
// grid (T/64, H), 128 threads (4 waves); wave w owns 16 queries.
__global__ __launch_bounds__(128) void attn_kernel(const __bf16* __restrict__ Q,
                                                   const __bf16* __restrict__ Kb,
                                                   const __bf16* __restrict__ Vb,
                                                   __bf16* __restrict__ O) {
  __shared__ __bf16 Ks[32][128];     // K tile, row-major [key][d]  (TDM staged)
  __shared__ __bf16 Vt[128][32];     // V tile transposed [d][key]
  __shared__ __bf16 Pl[4][16][32];   // per-wave P staging (C-layout -> A-layout)
  const int h = blockIdx.y, kvh = h >> 2;
  const int qb0 = blockIdx.x * 64;
  const int tid = threadIdx.x, wv = tid >> 5, lane = tid & 31;
  const int hf = lane >> 4, idx = lane & 15;
  const int qb = qb0 + wv * 16;
  v16bf qf[4];
  {
    const __bf16* qp = Q + ((size_t)(qb + idx) * Hh + h) * HDd;
#pragma unroll
    for (int ks = 0; ks < 4; ++ks) {
      v8bf lo = ld8(qp + ks * 32 + hf * 8);
      v8bf hi = ld8(qp + ks * 32 + 16 + hf * 8);
#pragma unroll
      for (int i = 0; i < 8; ++i) { qf[ks][i] = lo[i]; qf[ks][i + 8] = hi[i]; }
    }
  }
  float mrow[8], lrow[8];
  v8f oacc[8];
#pragma unroll
  for (int i = 0; i < 8; ++i) { mrow[i] = -3.0e38f; lrow[i] = 0.f; v8f z = {}; oacc[i] = z; }
  const float scale = 0.08838834764831845f;  // 1/sqrt(128)
  for (int kb = 0; kb <= qb0 + 63; kb += 32) {
    __syncthreads();
#if USE_TDM
    if (wv == 0)
      tdm_load_2d(lds_offset(&Ks[0][0]),
                  Kb + ((size_t)kb * KVHh + kvh) * HDd,
                  (unsigned int)HDd, (unsigned long long)(KVHh * HDd), 128u, 32u);
#endif
    {  // V tile, written transposed (and K tile when TDM unavailable)
      int row = tid >> 2, off = (tid & 3) * 32;
#if !USE_TDM
      const __bf16* kp = Kb + ((size_t)(kb + row) * KVHh + kvh) * HDd + off;
#endif
      const __bf16* vp = Vb + ((size_t)(kb + row) * KVHh + kvh) * HDd + off;
#pragma unroll
      for (int j = 0; j < 4; ++j) {
#if !USE_TDM
        *reinterpret_cast<v8bf*>(&Ks[row][off + j * 8]) = ld8(kp + j * 8);
#endif
        v8bf vvv = ld8(vp + j * 8);
#pragma unroll
        for (int e = 0; e < 8; ++e) Vt[off + j * 8 + e][row] = vvv[e];
      }
    }
#if USE_TDM
    if (wv == 0) __builtin_amdgcn_s_wait_tensorcnt(0);
#endif
    __syncthreads();
    if (kb > qb + 15) continue;  // wave-uniform; causal: nothing to do
    // S = Q @ K^T  (B frag: lane = key col, elems 0-7 = d hf*8.., 8-15 = d 16+hf*8.. -> Ks row read)
    v8f s0 = {}, s1 = {};
#pragma unroll
    for (int ks = 0; ks < 4; ++ks) {
      v16bf kf0, kf1;
      const __bf16* pk = &Ks[idx][ks * 32 + hf * 8];
      v8bf lo = ld8(pk), hi = ld8(pk + 16);
#pragma unroll
      for (int i = 0; i < 8; ++i) { kf0[i] = lo[i]; kf0[i + 8] = hi[i]; }
      pk = &Ks[16 + idx][ks * 32 + hf * 8];
      lo = ld8(pk); hi = ld8(pk + 16);
#pragma unroll
      for (int i = 0; i < 8; ++i) { kf1[i] = lo[i]; kf1[i + 8] = hi[i]; }
      s0 = wmma_bf16(qf[ks], kf0, s0);
      s1 = wmma_bf16(qf[ks], kf1, s1);
    }
    // online softmax: row of elem v is q = qb + v + hf*8; cols across 16 lanes
#pragma unroll
    for (int v = 0; v < 8; ++v) {
      int q = qb + v + hf * 8;
      float a = (kb + idx      <= q) ? s0[v] * scale : -3.0e38f;
      float b = (kb + 16 + idx <= q) ? s1[v] * scale : -3.0e38f;
      float mx = rmax_half(fmaxf(a, b));
      float mnew = fmaxf(mrow[v], mx);
      float alpha = __expf(mrow[v] - mnew);
      float p0 = __expf(a - mnew);
      float p1 = __expf(b - mnew);
      float sum = rsum_half(p0 + p1);
      lrow[v] = lrow[v] * alpha + sum;
      mrow[v] = mnew;
#pragma unroll
      for (int dt = 0; dt < 8; ++dt) oacc[dt][v] *= alpha;
      Pl[wv][v + hf * 8][idx]      = (__bf16)p0;
      Pl[wv][v + hf * 8][16 + idx] = (__bf16)p1;
    }
    // re-read P as A-frag (same-wave LDS ops are in-order)
    v16bf pf;
    {
      const __bf16* pp = &Pl[wv][idx][hf * 8];
      v8bf lo = ld8(pp), hi = ld8(pp + 16);
#pragma unroll
      for (int i = 0; i < 8; ++i) { pf[i] = lo[i]; pf[i + 8] = hi[i]; }
    }
    // O += P @ V : B frag lane = d col, elems 0-7 = key hf*8.., 8-15 = key 16+hf*8.. -> Vt row read
#pragma unroll
    for (int dt = 0; dt < 8; ++dt) {
      v16bf vf;
      const __bf16* pv = &Vt[dt * 16 + idx][hf * 8];
      v8bf lo = ld8(pv), hi = ld8(pv + 16);
#pragma unroll
      for (int i = 0; i < 8; ++i) { vf[i] = lo[i]; vf[i + 8] = hi[i]; }
      oacc[dt] = wmma_bf16(pf, vf, oacc[dt]);
    }
  }
#pragma unroll
  for (int dt = 0; dt < 8; ++dt)
#pragma unroll
    for (int v = 0; v < 8; ++v) {
      int q = qb + v + hf * 8;
      float val = oacc[dt][v] / lrow[v];
      O[(size_t)q * (Hh * HDd) + h * HDd + dt * 16 + idx] = (__bf16)val;
    }
}

__global__ __launch_bounds__(256) void router_kernel(const __bf16* __restrict__ x,
                                                     const float* __restrict__ gw,
                                                     float* __restrict__ comb) {
  __shared__ float part[256][8];
  int t = blockIdx.x;
  float acc[8];
#pragma unroll
  for (int e = 0; e < 8; ++e) acc[e] = 0.f;
  for (int d = threadIdx.x; d < Dd; d += 256) {
    float xv = (float)x[(size_t)t * Dd + d];
#pragma unroll
    for (int e = 0; e < 8; ++e) acc[e] += xv * gw[(size_t)d * Ee + e];
  }
#pragma unroll
  for (int e = 0; e < 8; ++e) part[threadIdx.x][e] = acc[e];
  __syncthreads();
  for (int s = 128; s > 0; s >>= 1) {
    if (threadIdx.x < s)
#pragma unroll
      for (int e = 0; e < 8; ++e) part[threadIdx.x][e] += part[threadIdx.x + s][e];
    __syncthreads();
  }
  if (threadIdx.x == 0) {
    float lg[8], pr[8];
    float mx = -3.0e38f;
    for (int e = 0; e < 8; ++e) { lg[e] = part[0][e]; mx = fmaxf(mx, lg[e]); }
    float sum = 0.f;
    for (int e = 0; e < 8; ++e) { pr[e] = __expf(lg[e] - mx); sum += pr[e]; }
    for (int e = 0; e < 8; ++e) pr[e] /= sum;
    int i0 = 0;
    for (int e = 1; e < 8; ++e) if (pr[e] > pr[i0]) i0 = e;
    int i1 = (i0 == 0) ? 1 : 0;
    for (int e = 0; e < 8; ++e) if (e != i0 && pr[e] > pr[i1]) i1 = e;
    float tot = pr[i0] + pr[i1];
    for (int e = 0; e < 8; ++e) comb[(size_t)t * Ee + e] = 0.f;
    comb[(size_t)t * Ee + i0] = pr[i0] / tot;
    comb[(size_t)t * Ee + i1] = pr[i1] / tot;
  }
}

__global__ void silu_mul_kernel(const float* __restrict__ g, const float* __restrict__ u,
                                __bf16* __restrict__ o, long n) {
  long i = (long)blockIdx.x * blockDim.x + threadIdx.x;
  long stride = (long)gridDim.x * blockDim.x;
  for (; i < n; i += stride) {
    float gv = g[i];
    float s = gv / (1.f + __expf(-gv));
    o[i] = (__bf16)(s * u[i]);
  }
}

__global__ void moe_acc_kernel(float* __restrict__ out, const float* __restrict__ y,
                               const float* __restrict__ comb, const float* __restrict__ h,
                               int e, int first) {
  long n = (long)Td * Dd;
  long i = (long)blockIdx.x * blockDim.x + threadIdx.x;
  long stride = (long)gridDim.x * blockDim.x;
  for (; i < n; i += stride) {
    int t = (int)(i >> 11);  // Dd == 2048
    float w = comb[(size_t)t * Ee + e];
    float base = first ? h[i] : out[i];
    out[i] = base + w * y[i];
  }
}

extern "C" void kernel_launch(void* const* d_in, const int* in_sizes, int n_in,
                              void* d_out, int out_size, void* d_ws, size_t ws_size,
                              hipStream_t stream) {
  (void)in_sizes; (void)n_in; (void)out_size; (void)ws_size;
  const float* hidden = (const float*)d_in[0];
  const int*   pos    = (const int*)d_in[1];
  const float* ln1w   = (const float*)d_in[2];
  const float* ln2w   = (const float*)d_in[3];
  const float* wq     = (const float*)d_in[4];
  const float* bq     = (const float*)d_in[5];
  const float* wk     = (const float*)d_in[6];
  const float* bk     = (const float*)d_in[7];
  const float* wvp    = (const float*)d_in[8];
  const float* bvp    = (const float*)d_in[9];
  const float* wo     = (const float*)d_in[10];
  const float* gatew  = (const float*)d_in[11];
  const float* wgate  = (const float*)d_in[12];
  const float* wup    = (const float*)d_in[13];
  const float* wdown  = (const float*)d_in[14];
  float* out = (float*)d_out;

  char* ws = (char*)d_ws;
  size_t off = 0;
  auto alloc = [&](size_t elems, size_t elsz) -> void* {
    void* p = ws + off;
    off = (off + elems * elsz + 255) & ~(size_t)255;
    return p;
  };
  __bf16* xb    = (__bf16*)alloc((size_t)Td * Dd, 2);
  __bf16* wq_b  = (__bf16*)alloc((size_t)Dd * (Hh * HDd), 2);
  __bf16* wk_b  = (__bf16*)alloc((size_t)Dd * (KVHh * HDd), 2);
  __bf16* wv_b  = (__bf16*)alloc((size_t)Dd * (KVHh * HDd), 2);
  __bf16* wo_b  = (__bf16*)alloc((size_t)(Hh * HDd) * Dd, 2);
  __bf16* wg_b  = (__bf16*)alloc((size_t)Ee * Dd * Ii, 2);
  __bf16* wu_b  = (__bf16*)alloc((size_t)Ee * Dd * Ii, 2);
  __bf16* wd_b  = (__bf16*)alloc((size_t)Ee * Ii * Dd, 2);
  float*  qfb   = (float*)alloc((size_t)Td * Hh * HDd, 4);
  float*  kfb   = (float*)alloc((size_t)Td * KVHh * HDd, 4);
  float*  vfb   = (float*)alloc((size_t)Td * KVHh * HDd, 4);
  __bf16* qrb   = (__bf16*)alloc((size_t)Td * Hh * HDd, 2);
  __bf16* krb   = (__bf16*)alloc((size_t)Td * KVHh * HDd, 2);
  __bf16* vrb   = (__bf16*)alloc((size_t)Td * KVHh * HDd, 2);
  __bf16* attnb = (__bf16*)alloc((size_t)Td * Dd, 2);
  float*  hbuf  = (float*)alloc((size_t)Td * Dd, 4);
  __bf16* x2b   = (__bf16*)alloc((size_t)Td * Dd, 2);
  float*  comb  = (float*)alloc((size_t)Td * Ee, 4);
  float*  gbuf  = (float*)alloc((size_t)Td * Ii, 4);
  float*  ubuf  = (float*)alloc((size_t)Td * Ii, 4);
  __bf16* ffb   = (__bf16*)alloc((size_t)Td * Ii, 2);
  float*  ybuf  = (float*)alloc((size_t)Td * Dd, 4);

  auto cast = [&](const float* in, __bf16* o, long n) {
    cast_bf16_kernel<<<2048, 256, 0, stream>>>(in, o, n);
  };
  cast(wq,    wq_b, (long)Dd * Hh * HDd);
  cast(wk,    wk_b, (long)Dd * KVHh * HDd);
  cast(wvp,   wv_b, (long)Dd * KVHh * HDd);
  cast(wo,    wo_b, (long)Hh * HDd * Dd);
  cast(wgate, wg_b, (long)Ee * Dd * Ii);
  cast(wup,   wu_b, (long)Ee * Dd * Ii);
  cast(wdown, wd_b, (long)Ee * Ii * Dd);

  rmsnorm_bf16_kernel<<<Td, 256, 0, stream>>>(hidden, ln1w, xb);

  gemm_bf16_kernel<<<dim3((Hh * HDd) / 64, Td / 128), 256, 0, stream>>>(
      xb, wq_b, qfb, bq, nullptr, Hh * HDd, Dd);
  gemm_bf16_kernel<<<dim3((KVHh * HDd) / 64, Td / 128), 256, 0, stream>>>(
      xb, wk_b, kfb, bk, nullptr, KVHh * HDd, Dd);
  gemm_bf16_kernel<<<dim3((KVHh * HDd) / 64, Td / 128), 256, 0, stream>>>(
      xb, wv_b, vfb, bvp, nullptr, KVHh * HDd, Dd);

  {
    long tq = (long)Td * Hh * 64;
    rope_bf16_kernel<<<(int)((tq + 255) / 256), 256, 0, stream>>>(qfb, pos, qrb, Hh, tq);
    long tk = (long)Td * KVHh * 64;
    rope_bf16_kernel<<<(int)((tk + 255) / 256), 256, 0, stream>>>(kfb, pos, krb, KVHh, tk);
  }
  cast(vfb, vrb, (long)Td * KVHh * HDd);

  attn_kernel<<<dim3(Td / 64, Hh), 128, 0, stream>>>(qrb, krb, vrb, attnb);

  // h = hidden + attn @ wo   (residual fused in epilogue)
  gemm_bf16_kernel<<<dim3(Dd / 64, Td / 128), 256, 0, stream>>>(
      attnb, wo_b, hbuf, nullptr, hidden, Dd, Hh * HDd);

  rmsnorm_bf16_kernel<<<Td, 256, 0, stream>>>(hbuf, ln2w, x2b);
  router_kernel<<<Td, 256, 0, stream>>>(x2b, gatew, comb);

  for (int e = 0; e < Ee; ++e) {
    const __bf16* wg_e = wg_b + (size_t)e * Dd * Ii;
    const __bf16* wu_e = wu_b + (size_t)e * Dd * Ii;
    const __bf16* wd_e = wd_b + (size_t)e * Ii * Dd;
    gemm_bf16_kernel<<<dim3(Ii / 64, Td / 128), 256, 0, stream>>>(
        x2b, wg_e, gbuf, nullptr, nullptr, Ii, Dd);
    gemm_bf16_kernel<<<dim3(Ii / 64, Td / 128), 256, 0, stream>>>(
        x2b, wu_e, ubuf, nullptr, nullptr, Ii, Dd);
    silu_mul_kernel<<<2048, 256, 0, stream>>>(gbuf, ubuf, ffb, (long)Td * Ii);
    gemm_bf16_kernel<<<dim3(Dd / 64, Td / 128), 256, 0, stream>>>(
        ffb, wd_e, ybuf, nullptr, nullptr, Dd, Ii);
    moe_acc_kernel<<<4096, 256, 0, stream>>>(out, ybuf, comb, hbuf, e, e == 0 ? 1 : 0);
  }
}
